// NoisyTopExpertsPerItemRouter_13048110645658
// MI455X (gfx1250) — matrix operand order repaired and checked by reading, not compiled, not run on hardware
//
#include <hip/hip_runtime.h>

typedef __attribute__((ext_vector_type(16))) __bf16 v16bf;
typedef __attribute__((ext_vector_type(8)))  float  v8f;

constexpr int kH = 2048;   // hidden
constexpr int kE = 64;     // experts
constexpr int kN = 32768;  // G*S tokens

constexpr int OFF_IDX   = 0;
constexpr int OFF_W     = kN * 2;      // 65536
constexpr int OFF_AUX   = kN * 4;      // 131072
constexpr int OFF_PROBS = kN * 4 + 1;  // 131073

// A-fragment (16x32 bf16, MxK): lane holds row M=lane&15; with kb=8*(lane>>4),
// elements 0..7  = K kb..kb+7, elements 8..15 = K kb+16..kb+23.
// Caller passes p = &row[k0 + kb].
__device__ __forceinline__ v16bf load_a_frag(const float* __restrict__ p) {
  float4 x0 = ((const float4*)p)[0];
  float4 x1 = ((const float4*)p)[1];
  float4 y0 = ((const float4*)(p + 16))[0];
  float4 y1 = ((const float4*)(p + 16))[1];
  v16bf r;
  r[0] = (__bf16)x0.x; r[1] = (__bf16)x0.y; r[2]  = (__bf16)x0.z; r[3]  = (__bf16)x0.w;
  r[4] = (__bf16)x1.x; r[5] = (__bf16)x1.y; r[6]  = (__bf16)x1.z; r[7]  = (__bf16)x1.w;
  r[8] = (__bf16)y0.x; r[9] = (__bf16)y0.y; r[10] = (__bf16)y0.z; r[11] = (__bf16)y0.w;
  r[12] = (__bf16)y1.x; r[13] = (__bf16)y1.y; r[14] = (__bf16)y1.z; r[15] = (__bf16)y1.w;
  return r;
}

// B-fragment (32x16 bf16, KxN): lane holds column N=lane&15 (a row of W);
// with kb=8*(lane>>4), elements 0..7 = K kb..kb+7, elements 8..15 = K kb+16..kb+23.
// Caller passes p = &Wrow[k0 + kb].
__device__ __forceinline__ v16bf load_b_frag(const float* __restrict__ p) {
  float4 x0 = ((const float4*)p)[0];
  float4 x1 = ((const float4*)p)[1];
  float4 x2 = ((const float4*)(p + 16))[0];
  float4 x3 = ((const float4*)(p + 16))[1];
  v16bf r;
  r[0]  = (__bf16)x0.x; r[1]  = (__bf16)x0.y; r[2]  = (__bf16)x0.z; r[3]  = (__bf16)x0.w;
  r[4]  = (__bf16)x1.x; r[5]  = (__bf16)x1.y; r[6]  = (__bf16)x1.z; r[7]  = (__bf16)x1.w;
  r[8]  = (__bf16)x2.x; r[9]  = (__bf16)x2.y; r[10] = (__bf16)x2.z; r[11] = (__bf16)x2.w;
  r[12] = (__bf16)x3.x; r[13] = (__bf16)x3.y; r[14] = (__bf16)x3.z; r[15] = (__bf16)x3.w;
  return r;
}

#define WMMA_BF16(A, B, C) \
  __builtin_amdgcn_wmma_f32_16x16x32_bf16(false, (A), false, (B), (short)0, (C), false, false)

__global__ void init_ws_kernel(float* __restrict__ ws) {
  if (threadIdx.x < 128) ws[threadIdx.x] = 0.f;
}

__global__ __launch_bounds__(256) void router_gemm_kernel(
    const float* __restrict__ tokens, const float* __restrict__ noise,
    const float* __restrict__ Wm, const float* __restrict__ bias,
    float* __restrict__ out, float* __restrict__ ws) {
  const int lane = (int)(threadIdx.x & 31);
  const int ln = lane & 15;    // column / lane-in-half
  const int lh = lane >> 4;    // lane half
  const int wave = (int)((blockIdx.x * 256u + threadIdx.x) >> 5);
  const int tok0 = wave * 32;  // 32 tokens per wave

  const float* aP0 = tokens + (size_t)(tok0 + ln) * kH + lh * 8;
  const float* aP1 = aP0 + (size_t)16 * kH;
  const float* bP  = Wm + (size_t)ln * kH + lh * 8;

  v8f acc[2][4] = {};  // 2 token-subtiles x 4 expert-subtiles

  for (int k0 = 0; k0 < kH; k0 += 32) {
    v16bf a0 = load_a_frag(aP0 + k0);
    v16bf a1 = load_a_frag(aP1 + k0);
    v16bf b0 = load_b_frag(bP + k0);
    v16bf b1 = load_b_frag(bP + 16 * kH + k0);
    v16bf b2 = load_b_frag(bP + 32 * kH + k0);
    v16bf b3 = load_b_frag(bP + 48 * kH + k0);
    acc[0][0] = WMMA_BF16(a0, b0, acc[0][0]);
    acc[0][1] = WMMA_BF16(a0, b1, acc[0][1]);
    acc[0][2] = WMMA_BF16(a0, b2, acc[0][2]);
    acc[0][3] = WMMA_BF16(a0, b3, acc[0][3]);
    acc[1][0] = WMMA_BF16(a1, b0, acc[1][0]);
    acc[1][1] = WMMA_BF16(a1, b1, acc[1][1]);
    acc[1][2] = WMMA_BF16(a1, b2, acc[1][2]);
    acc[1][3] = WMMA_BF16(a1, b3, acc[1][3]);
  }

  float bv[4];
  #pragma unroll
  for (int t = 0; t < 4; ++t) bv[t] = bias[t * 16 + ln];

  float imp[4] = {0.f, 0.f, 0.f, 0.f};
  float lod[4] = {0.f, 0.f, 0.f, 0.f};

  #pragma unroll
  for (int mt = 0; mt < 2; ++mt) {
    #pragma unroll
    for (int r = 0; r < 8; ++r) {
      const int token = tok0 + mt * 16 + 8 * lh + r;  // C layout: M = r + 8*laneHalf
      float v[4];
      #pragma unroll
      for (int t = 0; t < 4; ++t)
        v[t] = acc[mt][t][r] + bv[t] + noise[(size_t)token * kE + t * 16 + ln];

      // softmax over 64 experts: 4 local values x 16 lanes (within lane half)
      float m = fmaxf(fmaxf(v[0], v[1]), fmaxf(v[2], v[3]));
      #pragma unroll
      for (int off = 8; off >= 1; off >>= 1) m = fmaxf(m, __shfl_xor(m, off, 32));
      float e[4], s = 0.f;
      #pragma unroll
      for (int t = 0; t < 4; ++t) { e[t] = __expf(v[t] - m); s += e[t]; }
      #pragma unroll
      for (int off = 8; off >= 1; off >>= 1) s += __shfl_xor(s, off, 32);
      const float inv = 1.f / s;

      float p[4];
      #pragma unroll
      for (int t = 0; t < 4; ++t) {
        p[t] = e[t] * inv;
        __builtin_nontemporal_store(p[t], &out[OFF_PROBS + (size_t)token * kE + t * 16 + ln]);
        imp[t] += p[t];
        lod[t] += (p[t] > 0.f) ? 1.f : 0.f;
      }

      // per-lane top-2 (index tie-break: lower index wins, like lax.top_k)
      float v1 = -3.4e38f, v2 = -3.4e38f; int i1 = 0x7fffffff, i2 = 0x7fffffff;
      #pragma unroll
      for (int t = 0; t < 4; ++t) {
        float pv = p[t]; int pi = t * 16 + ln;
        if (pv > v1 || (pv == v1 && pi < i1)) { v2 = v1; i2 = i1; v1 = pv; i1 = pi; }
        else if (pv > v2 || (pv == v2 && pi < i2)) { v2 = pv; i2 = pi; }
      }
      // butterfly merge across the 16-lane half
      #pragma unroll
      for (int off = 8; off >= 1; off >>= 1) {
        float w1 = __shfl_xor(v1, off, 32), w2 = __shfl_xor(v2, off, 32);
        int   j1 = __shfl_xor(i1, off, 32), j2 = __shfl_xor(i2, off, 32);
        bool aF = (v1 > w1) || (v1 == w1 && i1 < j1);
        float n1  = aF ? v1 : w1; int m1  = aF ? i1 : j1;
        float c2v = aF ? v2 : w2; int c2i = aF ? i2 : j2;   // loser-side cand = winner's 2nd
        float o1v = aF ? w1 : v1; int o1i = aF ? j1 : i1;   // other side's 1st
        bool sF = (c2v > o1v) || (c2v == o1v && c2i < o1i);
        float n2 = sF ? c2v : o1v; int m2 = sF ? c2i : o1i;
        v1 = n1; i1 = m1; v2 = n2; i2 = m2;
      }
      if (ln == 0) {
        float invd = 1.f / (v1 + v2 + 1e-9f);
        out[OFF_IDX + token * 2 + 0] = (float)i1;
        out[OFF_IDX + token * 2 + 1] = (float)i2;
        out[OFF_W + token * 2 + 0] = v1 * invd;
        out[OFF_W + token * 2 + 1] = v2 * invd;
      }
    }
  }

  // fold the two lane-halves (same expert, different token subsets), then atomics
  #pragma unroll
  for (int t = 0; t < 4; ++t) {
    imp[t] += __shfl_xor(imp[t], 16, 32);
    lod[t] += __shfl_xor(lod[t], 16, 32);
  }
  if (lh == 0) {
    #pragma unroll
    for (int t = 0; t < 4; ++t) {
      atomicAdd(&ws[t * 16 + ln], imp[t]);
      atomicAdd(&ws[64 + t * 16 + ln], lod[t]);
    }
  }
}

__global__ void finalize_aux_kernel(const float* __restrict__ ws, float* __restrict__ out) {
  __shared__ float sx[kE];
  int e = (int)threadIdx.x;
  sx[e] = ws[e] * ws[64 + e];
  __syncthreads();
  if (e == 0) {
    float mean = 0.f;
    for (int i = 0; i < kE; ++i) mean += sx[i];
    mean *= (1.f / kE);
    float var = 0.f;
    for (int i = 0; i < kE; ++i) { float d = sx[i] - mean; var += d * d; }
    var *= (1.f / (kE - 1));     // unbiased (torch .var default)
    out[OFF_AUX] = var * 0.01f;
  }
}

extern "C" void kernel_launch(void* const* d_in, const int* in_sizes, int n_in,
                              void* d_out, int out_size, void* d_ws, size_t ws_size,
                              hipStream_t stream) {
  const float* tokens = (const float*)d_in[0];
  const float* noise  = (const float*)d_in[1];
  const float* Wm     = (const float*)d_in[2];
  const float* bias   = (const float*)d_in[3];
  float* out = (float*)d_out;
  float* ws  = (float*)d_ws;

  hipLaunchKernelGGL(init_ws_kernel, dim3(1), dim3(128), 0, stream, ws);
  // 32 tokens/wave, 8 waves/block -> 256 tokens/block -> 128 blocks exactly covers 32768
  hipLaunchKernelGGL(router_gemm_kernel, dim3(kN / 256), dim3(256), 0, stream,
                     tokens, noise, Wm, bias, out, ws);
  hipLaunchKernelGGL(finalize_aux_kernel, dim3(1), dim3(kE), 0, stream, ws, out);
}
